// VanillaSFNN_2637109920465
// MI455X (gfx1250) — compile-verified, matrix operand order checked by reading, not run on hardware
//
#include <hip/hip_runtime.h>

typedef __attribute__((ext_vector_type(16))) _Float16 v16h;
typedef __attribute__((ext_vector_type(8)))  _Float16 v8h;
typedef __attribute__((ext_vector_type(4)))  _Float16 v4h;
typedef __attribute__((ext_vector_type(8)))  float    v8f;
typedef __attribute__((ext_vector_type(4)))  float    v4f;

#define B_ 64
#define T_ 4096
#define I_ 128
#define H_ 256
#define HCHUNK 64
#define NCHUNKS (H_ / HCHUNK)   // 4 H-chunks -> 256 blocks
#define TT 64                   // timesteps staged per iteration
#define XH_STRIDE 136           // f16 row stride (128 + pad) -> breaks 256B bank cycle
#define CUR_STRIDE 68           // f32 row stride (64 + pad)

// ---------------------------------------------------------------------------
// Fused kernel: cur = x*Wd^T (+bd)  ->  EMA scan over T  ->  partial dot w/ Wo
// One block per (batch b, H-chunk). 4 waves of 32 (wave32). Each wave owns a
// 16-wide N-tile of the H-chunk and computes cur via v_wmma_f32_16x16x32_f16.
// Global x loads for tile i+1 are software-pipelined under tile i's compute.
// ---------------------------------------------------------------------------
__global__ __launch_bounds__(128) void sfnn_fused(
    const float* __restrict__ x,   // [B,T,I]
    const float* __restrict__ Wd,  // [H,I]
    const float* __restrict__ bd,  // [H]
    const float* __restrict__ Wo,  // [O=1,H]
    const float* __restrict__ tau, // [H]
    float* __restrict__ ws)        // [B,T] partial accumulator (pre-zeroed)
{
    const int chunk = blockIdx.x;            // 0..3
    const int b     = blockIdx.y;            // 0..63
    const int tid   = threadIdx.x;           // 0..127
    const int wave  = tid >> 5;              // 0..3
    const int lane  = tid & 31;
    const int l16   = lane & 15;
    const int hi16  = (lane & 16) ? 1 : 0;   // upper half-wave flag
    const int hbase = chunk * HCHUNK;

    __shared__ _Float16 xh[TT * XH_STRIDE];      // staged x tile, f16
    __shared__ float    curLds[TT * CUR_STRIDE]; // cur tile (f32, post-WMMA)
    __shared__ float    outLds[TT * CUR_STRIDE]; // mem*Wo per (t,h)

    // ---- Build B fragments once: Wd chunk, f32 -> f16, ISA 16-bit B layout.
    // lane (0..15) -> N = wave*16 + lane; elems 0..7 = K klo..klo+7,
    // elems 8..15 = K klo+16..klo+23; upper half-wave offsets K by +8.
    v16h bfrag[4];
    {
        const int h = hbase + wave * 16 + l16;
        const float* wrow = Wd + (size_t)h * I_;
        #pragma unroll
        for (int kk = 0; kk < 4; ++kk) {
            const int klo = kk * 32 + hi16 * 8;
            v4f w0 = *(const v4f*)(wrow + klo);
            v4f w1 = *(const v4f*)(wrow + klo + 4);
            v4f w2 = *(const v4f*)(wrow + klo + 16);
            v4f w3 = *(const v4f*)(wrow + klo + 20);
            v16h f;
            #pragma unroll
            for (int j = 0; j < 4; ++j) {
                f[j]      = (_Float16)w0[j];
                f[j + 4]  = (_Float16)w1[j];
                f[j + 8]  = (_Float16)w2[j];
                f[j + 12] = (_Float16)w3[j];
            }
            bfrag[kk] = f;
        }
    }

    // ---- Per-channel scan parameters (one channel per thread 0..63) ----
    float alpha = 0.f, oma = 0.f, woh = 0.f, bdh = 0.f, mem = 0.f;
    if (tid < HCHUNK) {
        const int h = hbase + tid;
        const float tv = tau[h];
        alpha = 1.0f / (1.0f + __expf(-tv));   // sigmoid(tau)
        oma   = 1.0f - alpha;
        woh   = Wo[h];
        bdh   = bd[h];
    }

    const float* xb  = x  + ((size_t)b * T_) * I_;
    float*       wsb = ws + (size_t)b * T_;

    // ---- Prefetch tile 0 of x into registers (8 x 32B per thread) ----
    v4f pre[16];
    {
        const float* gx = xb;
        #pragma unroll
        for (int rep = 0; rep < 8; ++rep) {
            const int e = (rep * 128 + tid) * 8;
            pre[2 * rep]     = *(const v4f*)(gx + e);
            pre[2 * rep + 1] = *(const v4f*)(gx + e + 4);
        }
    }

    for (int t0 = 0; t0 < T_; t0 += TT) {
        __syncthreads();  // previous iteration's LDS consumers are done

        // ---- Convert prefetched x tile f32 -> f16 and stage into LDS ----
        #pragma unroll
        for (int rep = 0; rep < 8; ++rep) {
            const int e   = (rep * 128 + tid) * 8;
            const int row = e >> 7;
            const int col = e & 127;
            v4f a0 = pre[2 * rep];
            v4f a1 = pre[2 * rep + 1];
            v4h h0, h1;
            #pragma unroll
            for (int j = 0; j < 4; ++j) {
                h0[j] = (_Float16)a0[j];
                h1[j] = (_Float16)a1[j];
            }
            _Float16* dst = &xh[row * XH_STRIDE + col];
            *(v4h*)(dst)     = h0;
            *(v4h*)(dst + 4) = h1;
        }
        __syncthreads();

        // ---- Issue global loads for the NEXT tile; they retire under the
        //      WMMA + scan phases below (s_wait_loadcnt lands next iter) ----
        if (t0 + TT < T_) {
            const float* gx = xb + (size_t)(t0 + TT) * I_;
            #pragma unroll
            for (int rep = 0; rep < 8; ++rep) {
                const int e = (rep * 128 + tid) * 8;
                pre[2 * rep]     = *(const v4f*)(gx + e);
                pre[2 * rep + 1] = *(const v4f*)(gx + e + 4);
            }
        }

        // ---- WMMA: cur tile = xh(16xK) * bfrag(Kx16), K=128 in 4 steps.
        //      Batch all 8 A-fragment DS loads per M-tile before the 4
        //      chained WMMAs so DS latency overlaps matrix issue. ----
        #pragma unroll
        for (int mt = 0; mt < 4; ++mt) {
            const int trow = mt * 16 + l16;          // A: lane -> M row
            const _Float16* arow = &xh[trow * XH_STRIDE];
            v8h alo[4], ahi[4];
            #pragma unroll
            for (int kk = 0; kk < 4; ++kk) {
                const int klo = kk * 32 + hi16 * 8;  // A: half-wave K offset
                alo[kk] = *(const v8h*)(arow + klo);       // K klo..klo+7
                ahi[kk] = *(const v8h*)(arow + klo + 16);  // K klo+16..klo+23
            }
            v8f c = {};
            #pragma unroll
            for (int kk = 0; kk < 4; ++kk) {
                v16h a = __builtin_shufflevector(alo[kk], ahi[kk],
                         0,1,2,3,4,5,6,7,8,9,10,11,12,13,14,15);
                c = __builtin_amdgcn_wmma_f32_16x16x32_f16(
                        false, a, false, bfrag[kk], (short)0, c, false, false);
            }
            // C layout: VGPR r, lane -> (M = r + 8*hi16, N = l16)
            const int ncol = wave * 16 + l16;
            #pragma unroll
            for (int r = 0; r < 8; ++r) {
                const int tl = mt * 16 + r + hi16 * 8;
                curLds[tl * CUR_STRIDE + ncol] = c[r];
            }
        }
        __syncthreads();

        // ---- Serial EMA over the 64 staged steps (thread = channel) ----
        if (tid < HCHUNK) {
            #pragma unroll
            for (int tg = 0; tg < 8; ++tg) {
                float cs[8];
                #pragma unroll
                for (int j = 0; j < 8; ++j)   // batch LDS loads off the chain
                    cs[j] = curLds[(tg * 8 + j) * CUR_STRIDE + tid] + bdh;
                #pragma unroll
                for (int j = 0; j < 8; ++j) {
                    mem = __builtin_fmaf(alpha, mem, oma * cs[j]);
                    outLds[(tg * 8 + j) * CUR_STRIDE + tid] = mem * woh;
                }
            }
        }
        __syncthreads();

        // ---- Reduce 64 channels per timestep; accumulate across chunks ----
        if (tid < TT) {
            const float* row = &outLds[tid * CUR_STRIDE];
            v4f s = *(const v4f*)row;
            #pragma unroll
            for (int q = 1; q < 16; ++q) {
                v4f v = *(const v4f*)(row + q * 4);
                s.x += v.x; s.y += v.y; s.z += v.z; s.w += v.w;
            }
            atomicAdd(&wsb[t0 + tid], (s.x + s.y) + (s.z + s.w));
        }
    }
}

// ---------------------------------------------------------------------------
// Epilogue: out = sigmoid(acc + bo)
// ---------------------------------------------------------------------------
__global__ __launch_bounds__(256) void sfnn_out(const float* __restrict__ acc,
                                                const float* __restrict__ bo,
                                                float* __restrict__ out, int n)
{
    const int i = blockIdx.x * 256 + threadIdx.x;
    if (i < n) {
        const float z = acc[i] + bo[0];
        out[i] = 1.0f / (1.0f + __expf(-z));
    }
}

extern "C" void kernel_launch(void* const* d_in, const int* in_sizes, int n_in,
                              void* d_out, int out_size, void* d_ws, size_t ws_size,
                              hipStream_t stream)
{
    (void)in_sizes; (void)n_in; (void)out_size; (void)ws_size;
    const float* x   = (const float*)d_in[0];
    const float* Wd  = (const float*)d_in[1];
    const float* bd  = (const float*)d_in[2];
    const float* Wo  = (const float*)d_in[3];
    const float* bo  = (const float*)d_in[4];
    const float* tau = (const float*)d_in[5];
    float* out = (float*)d_out;
    float* acc = (float*)d_ws;   // [B,T] f32 partial accumulator (1 MB)

    hipMemsetAsync(acc, 0, (size_t)B_ * T_ * sizeof(float), stream);

    dim3 grid1(NCHUNKS, B_);     // 4 x 64 = 256 blocks
    sfnn_fused<<<grid1, 128, 0, stream>>>(x, Wd, bd, Wo, tau, acc);

    const int n = B_ * T_;
    sfnn_out<<<(n + 255) / 256, 256, 0, stream>>>(acc, bo, out, n);
}
